// Graphormer_13855564497049
// MI455X (gfx1250) — compile-verified, matrix operand order checked by reading, not run on hardware
//
#include <hip/hip_runtime.h>
#include <cmath>

// Problem constants (from reference)
#define NN    2048
#define GG    16
#define NGB   128
#define EE    32768
#define PP    131072
#define DD    128
#define HH    4
#define LL    2
#define MPDD  5
#define EDII  16
#define EDHH  64
#define INDD  64
#define OUTDD 64
#define FFNN  512
#define KPAD  32   // padded K for edge projection
#define LDSP  40   // LDS panel row stride in halfs (80B -> bank-spread, 16B aligned)

typedef _Float16 v16h __attribute__((ext_vector_type(16)));
typedef _Float16 v8h  __attribute__((ext_vector_type(8)));
typedef float    v8f  __attribute__((ext_vector_type(8)));

// ---------------------------------------------------------------- utilities

__global__ void k_zero(float* p, int n) {
  int i = blockIdx.x * blockDim.x + threadIdx.x;
  if (i < n) p[i] = 0.f;
}

__global__ void k_cvt(const float* __restrict__ src, _Float16* __restrict__ dst, int n) {
  int i = blockIdx.x * blockDim.x + threadIdx.x;
  if (i < n) dst[i] = (_Float16)src[i];
}

__global__ void k_degrees(const int* __restrict__ ei, int* outdeg, int* indeg) {
  int e = blockIdx.x * blockDim.x + threadIdx.x;
  if (e < EE) {
    atomicAdd(&outdeg[ei[e]], 1);       // edge_index[0] = src -> out_deg
    atomicAdd(&indeg [ei[EE + e]], 1);  // edge_index[1] = dst -> in_deg
  }
}

__global__ void k_centrality(float* h, const int* __restrict__ indeg,
                             const int* __restrict__ outdeg,
                             const float* __restrict__ z_in,
                             const float* __restrict__ z_out) {
  int t = blockIdx.x * blockDim.x + threadIdx.x;
  if (t < NN * DD) {
    int n = t >> 7, d = t & 127;
    int di = min(indeg[n], 63), doo = min(outdeg[n], 63);
    h[t] += z_in[di * DD + d] + z_out[doo * DD + d];
  }
}

// Batched zero-padded transpose to f16: Bt[b][n][k] = (k<K) ? B[b][k][n] : 0
__global__ void k_transpose(const float* __restrict__ B, long sB,
                            _Float16* __restrict__ Bt, long sBt,
                            int K, int N, int Kpad) {
  int t = blockIdx.x * blockDim.x + threadIdx.x;
  if (t >= N * Kpad) return;
  int n = t / Kpad, k = t - n * Kpad;
  long b = blockIdx.y;
  Bt[b * sBt + t] = (k < K) ? (_Float16)B[b * sB + (long)k * N + n] : (_Float16)0.f;
}

// Row pad to f16: dst[r][0..cd) = src[r][0..cs) then zeros
__global__ void k_padrows(const float* __restrict__ src, _Float16* __restrict__ dst,
                          int rows, int cs, int cd) {
  int t = blockIdx.x * blockDim.x + threadIdx.x;
  if (t < rows * cd) {
    int rr = t / cd, cc = t - rr * cd;
    dst[t] = (cc < cs) ? (_Float16)src[(long)rr * cs + cc] : (_Float16)0.f;
  }
}

__device__ __forceinline__ float geluf(float x) {
  return 0.5f * x * (1.f + erff(x * 0.70710678118654752f));
}

// async 16-byte global->LDS copy (gfx1250 GLOBAL_LOAD_ASYNC_TO_LDS_B128)
__device__ __forceinline__ void async16(unsigned lds_off, const _Float16* g) {
  asm volatile("global_load_async_to_lds_b128 %0, %1, off"
               :: "v"(lds_off), "v"(g) : "memory");
}
__device__ __forceinline__ void wait_async4() {
  asm volatile("s_wait_asynccnt 0x4" ::: "memory");
}
__device__ __forceinline__ void wait_async0() {
  asm volatile("s_wait_asynccnt 0x0" ::: "memory");
}

// ------------------------------------------------ cooperative WMMA GEMM
// C[b] = act( (A[b] @ B[b]) * alpha + bias[b] ) + resid[b]
// A: f16 [M,K] (lda). Bt: f16 [N,K] (ldb) = B transposed. Requires M%64==0,
// N%64==0, K%32==0. Block = 4 waves = 64x64 macro-tile; 64x32 A/B panels
// double-buffered in LDS via async loads; fragments via ds_load_b128.

template <typename CT>
__global__ __launch_bounds__(128)
void k_gemm(const _Float16* __restrict__ A, long sA, int lda,
            const _Float16* __restrict__ Bt, long sB, int ldb,
            const float* __restrict__ bias, long sBias,
            const float* resid, long sResid, int ldr,
            CT* C, long sC, int ldc,
            int K, float alpha, int act, int tn)
{
  __shared__ _Float16 smA[2][64 * LDSP];
  __shared__ _Float16 smB[2][64 * LDSP];

  int tile = blockIdx.x;
  int tmi = tile / tn, tni = tile - tmi * tn;
  long batch = blockIdx.y;
  A  += batch * sA + (long)(tmi * 64) * lda;
  Bt += batch * sB + (long)(tni * 64) * ldb;
  if (bias)  bias  += batch * sBias;
  if (resid) resid += batch * sResid;
  C  += batch * sC;

  int t = threadIdx.y * 32 + threadIdx.x;    // 0..127
  int srow0 = t >> 2, srow1 = srow0 + 32;    // staging rows
  int seg8  = (t & 3) << 3;                  // 8-half segment within 32-half panel row

  auto stage = [&](int buf, int k0) {
    async16((unsigned)(uintptr_t)&smA[buf][srow0 * LDSP + seg8], A + (long)srow0 * lda + k0 + seg8);
    async16((unsigned)(uintptr_t)&smA[buf][srow1 * LDSP + seg8], A + (long)srow1 * lda + k0 + seg8);
    async16((unsigned)(uintptr_t)&smB[buf][srow0 * LDSP + seg8], Bt + (long)srow0 * ldb + k0 + seg8);
    async16((unsigned)(uintptr_t)&smB[buf][srow1 * LDSP + seg8], Bt + (long)srow1 * ldb + k0 + seg8);
  };

  int lane = threadIdx.x;
  int r  = lane & 15;
  int hb = lane >> 4;                        // half-wave K-packing select
  int wy = threadIdx.y >> 1, wx = threadIdx.y & 1;  // wave position in 2x2

  v8f acc00 = {}, acc01 = {}, acc10 = {}, acc11 = {};
  int nsteps = K >> 5;
  stage(0, 0);
  for (int s = 0; s < nsteps; ++s) {
    int cur = s & 1;
    if (s + 1 < nsteps) { stage(cur ^ 1, (s + 1) << 5); wait_async4(); }
    else                { wait_async0(); }
    __syncthreads();                         // panel[cur] visible to all waves

    const _Float16* pa = &smA[cur][(wy * 32 + r) * LDSP + (hb << 3)];
    const _Float16* pb = &smB[cur][(wx * 32 + r) * LDSP + (hb << 3)];
    v8h aL0 = *(const v8h*)(pa);
    v8h aH0 = *(const v8h*)(pa + 16);
    v8h aL1 = *(const v8h*)(pa + 16 * LDSP);
    v8h aH1 = *(const v8h*)(pa + 16 * LDSP + 16);
    v8h bL0 = *(const v8h*)(pb);
    v8h bH0 = *(const v8h*)(pb + 16);
    v8h bL1 = *(const v8h*)(pb + 16 * LDSP);
    v8h bH1 = *(const v8h*)(pb + 16 * LDSP + 16);
    v16h af0 = __builtin_shufflevector(aL0, aH0, 0,1,2,3,4,5,6,7,8,9,10,11,12,13,14,15);
    v16h af1 = __builtin_shufflevector(aL1, aH1, 0,1,2,3,4,5,6,7,8,9,10,11,12,13,14,15);
    v16h bf0 = __builtin_shufflevector(bL0, bH0, 0,1,2,3,4,5,6,7,8,9,10,11,12,13,14,15);
    v16h bf1 = __builtin_shufflevector(bL1, bH1, 0,1,2,3,4,5,6,7,8,9,10,11,12,13,14,15);

    acc00 = __builtin_amdgcn_wmma_f32_16x16x32_f16(false, af0, false, bf0, (short)0, acc00, false, false);
    acc01 = __builtin_amdgcn_wmma_f32_16x16x32_f16(false, af0, false, bf1, (short)0, acc01, false, false);
    acc10 = __builtin_amdgcn_wmma_f32_16x16x32_f16(false, af1, false, bf0, (short)0, acc10, false, false);
    acc11 = __builtin_amdgcn_wmma_f32_16x16x32_f16(false, af1, false, bf1, (short)0, acc11, false, false);
    __syncthreads();                         // all reads of panel[cur] done
  }

  int gr0 = tmi * 64 + wy * 32;
  int gc0 = tni * 64 + wx * 32;
  int n0 = gc0 + r, n1 = n0 + 16;
  float bias0 = 0.f, bias1 = 0.f;
  if (bias) { bias0 = bias[n0]; bias1 = bias[n1]; }

  auto emit = [&](int row, int col, float v, float bs) {
    float tv = v * alpha + bs;
    if (act)   tv = geluf(tv);
    if (resid) tv += resid[(long)row * ldr + col];
    C[(long)row * ldc + col] = (CT)tv;
  };
#pragma unroll
  for (int i = 0; i < 8; ++i) {
    int row0 = gr0 + (hb << 3) + i;          // lanes 0-15 rows 0-7, lanes 16-31 rows 8-15
    int row1 = row0 + 16;
    emit(row0, n0, acc00[i], bias0);
    emit(row0, n1, acc01[i], bias1);
    emit(row1, n0, acc10[i], bias0);
    emit(row1, n1, acc11[i], bias1);
  }
}

// --------------------------------------------------------------- layernorm
__global__ __launch_bounds__(128)
void k_ln(const float* __restrict__ x, const float* __restrict__ g,
          const float* __restrict__ b, _Float16* __restrict__ y, int rows)
{
  int row  = blockIdx.x * 4 + (threadIdx.x >> 5);  // wave-uniform
  int lane = threadIdx.x & 31;
  if (row >= rows) return;
  const float* xr = x + (long)row * DD;
  float v0[4], s = 0.f, sq = 0.f;
#pragma unroll
  for (int j = 0; j < 4; ++j) {
    float t = xr[lane + 32 * j];
    v0[j] = t; s += t; sq += t * t;
  }
#pragma unroll
  for (int off = 16; off >= 1; off >>= 1) {
    s  += __shfl_xor(s,  off, 32);
    sq += __shfl_xor(sq, off, 32);
  }
  float mean = s * (1.f / DD);
  float var  = sq * (1.f / DD) - mean * mean;
  float rstd = rsqrtf(var + 1e-5f);
#pragma unroll
  for (int j = 0; j < 4; ++j) {
    int c = lane + 32 * j;
    y[(long)row * DD + c] = (_Float16)((v0[j] - mean) * rstd * g[c] + b[c]);
  }
}

// --------------------------------------------------- softmax (per 128-row)
// Off-block masked entries are exactly -0.0 in the reference -> each row gets
// (N-NG)*exp(-max) extra denominator mass, max clamped to >= 0.
__global__ __launch_bounds__(128)
void k_softmax(float* __restrict__ S, _Float16* __restrict__ Sh, int rows)
{
  int row  = blockIdx.x * 4 + (threadIdx.x >> 5);
  int lane = threadIdx.x & 31;
  if (row >= rows) return;
  float* sr = S + (long)row * NGB;
  _Float16* shr = Sh + (long)row * NGB;
  float v0[4], mx = -1e30f;
#pragma unroll
  for (int j = 0; j < 4; ++j) { v0[j] = sr[lane + 32 * j]; mx = fmaxf(mx, v0[j]); }
#pragma unroll
  for (int off = 16; off >= 1; off >>= 1) mx = fmaxf(mx, __shfl_xor(mx, off, 32));
  mx = fmaxf(mx, 0.f);
  float sum = 0.f;
#pragma unroll
  for (int j = 0; j < 4; ++j) { v0[j] = __expf(v0[j] - mx); sum += v0[j]; }
#pragma unroll
  for (int off = 16; off >= 1; off >>= 1) sum += __shfl_xor(sum, off, 32);
  sum += (float)(NN - NGB) * __expf(-mx);
  float inv = 1.f / sum;
#pragma unroll
  for (int j = 0; j < 4; ++j) {
    float p = v0[j] * inv;
    sr[lane + 32 * j]  = p;
    shr[lane + 32 * j] = (_Float16)p;
  }
}

// ---------------------- per-path edge encoding -> per-head bias scatter
__global__ void k_edgebias(const float* __restrict__ ea,
                           const float* __restrict__ ev,   // [H,MPD,EDH] of layer l
                           const float* __restrict__ b_spatial,
                           const int* __restrict__ p_src, const int* __restrict__ p_dst,
                           const int* __restrict__ p_edges, const int* __restrict__ p_lens,
                           float* __restrict__ Bb)
{
  int p = blockIdx.x * blockDim.x + threadIdx.x;
  if (p >= PP) return;
  int len = p_lens[p]; len = (len > MPDD) ? MPDD : len;
  float acc[HH] = {0.f, 0.f, 0.f, 0.f};
  for (int mstep = 0; mstep < len; ++mstep) {
    const float* er = ea + (long)p_edges[p * MPDD + mstep] * EDHH;
    for (int e = 0; e < EDHH; e += 4) {
      float4 a = *(const float4*)(er + e);
#pragma unroll
      for (int hh = 0; hh < HH; ++hh) {
        float4 w = *(const float4*)(ev + (hh * MPDD + mstep) * EDHH + e);
        acc[hh] += w.x * a.x + w.y * a.y + w.z * a.z + w.w * a.w;
      }
    }
  }
  float invl = 1.f / (float)len;
  int src = p_src[p], dst = p_dst[p];
  int gidx = src / NGB;
  int ii = src - gidx * NGB, jj = dst - gidx * NGB;
  float bsp = b_spatial[len - 1];
#pragma unroll
  for (int hh = 0; hh < HH; ++hh)
    Bb[(((long)hh * GG + gidx) * NGB + ii) * NGB + jj] = acc[hh] * invl + bsp;
}

// ---------------------------------------------------- head-concat transpose
__global__ void k_concat(const float* __restrict__ o, _Float16* __restrict__ ocat) {
  int t = blockIdx.x * blockDim.x + threadIdx.x;
  if (t < NN * HH * DD) {
    int n = t >> 9, rem = t & 511, hh = rem >> 7, d = rem & 127;
    ocat[t] = (_Float16)o[((long)hh * NN + n) * DD + d];
  }
}

// ------------------------------------------------------------------ driver

template <typename CT>
static void gemm(hipStream_t s,
                 const _Float16* A, long sA, int lda,
                 const _Float16* Bt, long sB, int ldb,
                 const float* bias, long sBias,
                 const float* resid, long sResid, int ldr,
                 CT* C, long sC, int ldc,
                 int M, int Nn, int K, float alpha, int act, int batches)
{
  int tm = M / 64, tn = Nn / 64;
  dim3 grid(tm * tn, batches);
  dim3 block(32, 4);
  k_gemm<CT><<<grid, block, 0, s>>>(A, sA, lda, Bt, sB, ldb,
                                    bias, sBias, resid, sResid, ldr,
                                    C, sC, ldc, K, alpha, act, tn);
}

static void transpose(hipStream_t s, const float* B, long sB, _Float16* Bt, long sBt,
                      int K, int N, int Kpad, int batches) {
  dim3 grid((N * Kpad + 255) / 256, batches);
  k_transpose<<<grid, 256, 0, s>>>(B, sB, Bt, sBt, K, N, Kpad);
}

extern "C" void kernel_launch(void* const* d_in, const int* in_sizes, int n_in,
                              void* d_out, int out_size, void* d_ws, size_t ws_size,
                              hipStream_t stream) {
  const float* x         = (const float*)d_in[0];
  const float* edge_attr = (const float*)d_in[1];
  const int*   edge_idx  = (const int*)  d_in[2];
  const int*   path_src  = (const int*)  d_in[3];
  const int*   path_dst  = (const int*)  d_in[4];
  const int*   path_edges= (const int*)  d_in[5];
  const int*   path_lens = (const int*)  d_in[6];
  /* d_in[7] = ptr (block structure implicit) */
  const float* W_in  = (const float*)d_in[8];
  const float* b_in  = (const float*)d_in[9];
  const float* W_ein = (const float*)d_in[10];
  const float* b_ein = (const float*)d_in[11];
  const float* z_in  = (const float*)d_in[12];
  const float* z_out = (const float*)d_in[13];
  const float* b_sp  = (const float*)d_in[14];
  const float* ln1_g = (const float*)d_in[15];
  const float* ln1_b = (const float*)d_in[16];
  const float* Wq    = (const float*)d_in[17];
  const float* bq    = (const float*)d_in[18];
  const float* Wk    = (const float*)d_in[19];
  const float* bk    = (const float*)d_in[20];
  const float* Wv    = (const float*)d_in[21];
  const float* bv    = (const float*)d_in[22];
  const float* evec  = (const float*)d_in[23];
  const float* Wo    = (const float*)d_in[24];
  const float* bo    = (const float*)d_in[25];
  const float* ln2_g = (const float*)d_in[26];
  const float* ln2_b = (const float*)d_in[27];
  const float* W_f1  = (const float*)d_in[28];
  const float* b_f1  = (const float*)d_in[29];
  const float* W_f2  = (const float*)d_in[30];
  const float* b_f2  = (const float*)d_in[31];
  const float* W_out = (const float*)d_in[32];
  const float* b_out = (const float*)d_in[33];
  float* out = (float*)d_out;

  // workspace bump allocator (256B aligned)
  char* base = (char*)d_ws;
  size_t woff = 0;
  auto alloc = [&](size_t bytes) -> void* {
    woff = (woff + 255) & ~(size_t)255;
    void* p = base + woff;
    woff += bytes;
    return p;
  };
  float*    hbuf  = (float*)   alloc(sizeof(float)    * NN * DD);
  float*    eab   = (float*)   alloc(sizeof(float)    * (size_t)EE * EDHH);
  float*    vb    = (float*)   alloc(sizeof(float)    * (size_t)HH * NN * DD);
  float*    Sb    = (float*)   alloc(sizeof(float)    * (size_t)HH * NN * NGB);
  float*    Bb    = (float*)   alloc(sizeof(float)    * (size_t)HH * NN * NGB);
  float*    ob    = (float*)   alloc(sizeof(float)    * (size_t)HH * NN * DD);
  _Float16* xh    = (_Float16*)alloc(sizeof(_Float16) * NN * INDD);
  _Float16* xn    = (_Float16*)alloc(sizeof(_Float16) * NN * DD);
  _Float16* hh16  = (_Float16*)alloc(sizeof(_Float16) * NN * DD);
  _Float16* eap   = (_Float16*)alloc(sizeof(_Float16) * (size_t)EE * KPAD);
  _Float16* qh    = (_Float16*)alloc(sizeof(_Float16) * (size_t)HH * NN * DD);
  _Float16* kh    = (_Float16*)alloc(sizeof(_Float16) * (size_t)HH * NN * DD);
  _Float16* vT    = (_Float16*)alloc(sizeof(_Float16) * (size_t)HH * NN * DD);
  _Float16* Sh    = (_Float16*)alloc(sizeof(_Float16) * (size_t)HH * NN * NGB);
  _Float16* ocat  = (_Float16*)alloc(sizeof(_Float16) * (size_t)NN * HH * DD);
  _Float16* ffn   = (_Float16*)alloc(sizeof(_Float16) * (size_t)NN * FFNN);
  _Float16* WinT  = (_Float16*)alloc(sizeof(_Float16) * DD * INDD);
  _Float16* WeinT = (_Float16*)alloc(sizeof(_Float16) * EDHH * KPAD);
  _Float16* WqT   = (_Float16*)alloc(sizeof(_Float16) * (size_t)LL * HH * DD * DD);
  _Float16* WkT   = (_Float16*)alloc(sizeof(_Float16) * (size_t)LL * HH * DD * DD);
  _Float16* WvT   = (_Float16*)alloc(sizeof(_Float16) * (size_t)LL * HH * DD * DD);
  _Float16* WoT   = (_Float16*)alloc(sizeof(_Float16) * (size_t)LL * DD * HH * DD);
  _Float16* Wf1T  = (_Float16*)alloc(sizeof(_Float16) * (size_t)LL * FFNN * DD);
  _Float16* Wf2T  = (_Float16*)alloc(sizeof(_Float16) * (size_t)LL * DD * FFNN);
  _Float16* WoutT = (_Float16*)alloc(sizeof(_Float16) * OUTDD * DD);
  int*      degs  = (int*)     alloc(sizeof(int) * 2 * NN);
  int* outdeg = degs;
  int* indeg  = degs + NN;

  const float inv_sqrt_d = 1.0f / sqrtf((float)DD);

  // ---- weight pre-transposes to f16 K-contiguous layout ----
  transpose(stream, W_in, 0, WinT, 0, INDD, DD, INDD, 1);
  transpose(stream, W_ein, 0, WeinT, 0, EDII, EDHH, KPAD, 1);
  transpose(stream, Wq, (long)DD * DD, WqT, (long)DD * DD, DD, DD, DD, LL * HH);
  transpose(stream, Wk, (long)DD * DD, WkT, (long)DD * DD, DD, DD, DD, LL * HH);
  transpose(stream, Wv, (long)DD * DD, WvT, (long)DD * DD, DD, DD, DD, LL * HH);
  transpose(stream, Wo, (long)(HH * DD) * DD, WoT, (long)DD * (HH * DD), HH * DD, DD, HH * DD, LL);
  transpose(stream, W_f1, (long)DD * FFNN, Wf1T, (long)FFNN * DD, DD, FFNN, DD, LL);
  transpose(stream, W_f2, (long)FFNN * DD, Wf2T, (long)DD * FFNN, FFNN, DD, FFNN, LL);
  transpose(stream, W_out, 0, WoutT, 0, DD, OUTDD, DD, 1);
  k_padrows<<<(EE * KPAD + 255) / 256, 256, 0, stream>>>(edge_attr, eap, EE, EDII, KPAD);
  k_cvt<<<(NN * INDD + 255) / 256, 256, 0, stream>>>(x, xh, NN * INDD);

  // ---- degrees + input projections ----
  k_zero<<<(2 * NN + 255) / 256, 256, 0, stream>>>((float*)degs, 2 * NN);
  k_degrees<<<(EE + 255) / 256, 256, 0, stream>>>(edge_idx, outdeg, indeg);
  gemm<float>(stream, xh, 0, INDD, WinT, 0, INDD, b_in, 0, nullptr, 0, 0,
              hbuf, 0, DD, NN, DD, INDD, 1.f, 0, 1);
  k_centrality<<<(NN * DD + 255) / 256, 256, 0, stream>>>(hbuf, indeg, outdeg, z_in, z_out);
  gemm<float>(stream, eap, 0, KPAD, WeinT, 0, KPAD, b_ein, 0, nullptr, 0, 0,
              eab, 0, EDHH, EE, EDHH, KPAD, 1.f, 0, 1);

  for (int l = 0; l < LL; ++l) {
    // per-head spatial+edge-path bias blocks
    k_zero<<<((HH * NN * NGB) + 255) / 256, 256, 0, stream>>>(Bb, HH * NN * NGB);
    k_edgebias<<<(PP + 255) / 256, 256, 0, stream>>>(
        eab, evec + (size_t)l * HH * MPDD * EDHH, b_sp,
        path_src, path_dst, path_edges, path_lens, Bb);

    // LN1 (f16 out) + QKV batched over heads (q,k as f16; v as f32 for transpose)
    k_ln<<<(NN + 3) / 4, 128, 0, stream>>>(hbuf, ln1_g + l * DD, ln1_b + l * DD, xn, NN);
    gemm<_Float16>(stream, xn, 0, DD, WqT + (size_t)l * HH * DD * DD, (long)DD * DD, DD,
                   bq + (size_t)l * HH * DD, DD, nullptr, 0, 0,
                   qh, (long)NN * DD, DD, NN, DD, DD, 1.f, 0, HH);
    gemm<_Float16>(stream, xn, 0, DD, WkT + (size_t)l * HH * DD * DD, (long)DD * DD, DD,
                   bk + (size_t)l * HH * DD, DD, nullptr, 0, 0,
                   kh, (long)NN * DD, DD, NN, DD, DD, 1.f, 0, HH);
    gemm<float>(stream, xn, 0, DD, WvT + (size_t)l * HH * DD * DD, (long)DD * DD, DD,
                bv + (size_t)l * HH * DD, DD, nullptr, 0, 0,
                vb, (long)NN * DD, DD, NN, DD, DD, 1.f, 0, HH);

    // S = q @ k^T / sqrt(D) + bias   (batch = head x graph; diag blocks only)
    gemm<float>(stream, qh, (long)NGB * DD, DD,
                kh, (long)NGB * DD, DD,
                nullptr, 0, Bb, (long)NGB * NGB, NGB,
                Sb, (long)NGB * NGB, NGB,
                NGB, NGB, DD, inv_sqrt_d, 0, HH * GG);

    // softmax with off-block denominator correction (f32 + f16 copies)
    k_softmax<<<(HH * NN + 3) / 4, 128, 0, stream>>>(Sb, Sh, HH * NN);

    // transpose v per (h,g) -> f16 [D][NG], then o = probs @ v
    transpose(stream, vb, (long)NGB * DD, vT, (long)DD * NGB, NGB, DD, NGB, HH * GG);
    gemm<float>(stream, Sh, (long)NGB * NGB, NGB,
                vT, (long)DD * NGB, NGB,
                nullptr, 0, nullptr, 0, 0,
                ob, (long)NGB * DD, DD,
                NGB, DD, NGB, 1.f, 0, HH * GG);

    // concat heads (f16), output proj + residual (in-place into hbuf)
    k_concat<<<(NN * HH * DD + 255) / 256, 256, 0, stream>>>(ob, ocat);
    gemm<float>(stream, ocat, 0, HH * DD, WoT + (size_t)l * DD * HH * DD, 0, HH * DD,
                bo + l * DD, 0, hbuf, 0, DD,
                hbuf, 0, DD, NN, DD, HH * DD, 1.f, 0, 1);

    // LN2 + FFN (exact gelu, f16 intermediate) + residual
    k_ln<<<(NN + 3) / 4, 128, 0, stream>>>(hbuf, ln2_g + l * DD, ln2_b + l * DD, xn, NN);
    gemm<_Float16>(stream, xn, 0, DD, Wf1T + (size_t)l * FFNN * DD, 0, DD,
                   b_f1 + l * FFNN, 0, nullptr, 0, 0,
                   ffn, 0, FFNN, NN, FFNN, DD, 1.f, 1 /*gelu*/, 1);
    gemm<float>(stream, ffn, 0, FFNN, Wf2T + (size_t)l * DD * FFNN, 0, FFNN,
                b_f2 + l * DD, 0, hbuf, 0, DD,
                hbuf, 0, DD, NN, DD, FFNN, 1.f, 0, 1);
  }

  // output projection
  k_cvt<<<(NN * DD + 255) / 256, 256, 0, stream>>>(hbuf, hh16, NN * DD);
  gemm<float>(stream, hh16, 0, DD, WoutT, 0, DD, b_out, 0, nullptr, 0, 0,
              out, 0, OUTDD, NN, OUTDD, DD, 1.f, 0, 1);
}
